// VertexToEdgeLayer_46669114638610
// MI455X (gfx1250) — matrix operand order repaired and checked by reading, not compiled, not run on hardware
//
#include <hip/hip_runtime.h>
#include <hip/hip_fp16.h>

#define NUM_V 100000
#define NUM_E 400000
#define NNZ   800000
#define DM    128
#define KTOT  256        // concat K: msg(128) + h_e(128)
#define NGATE 512        // 4 * DM
#define LDA   264        // 8-half-padded LDS row stride (528B -> conflict-free b128)

typedef __attribute__((ext_vector_type(16))) _Float16 v16h;
typedef __attribute__((ext_vector_type(8)))  float    v8f;

union FragA { uint4 u[2]; v16h v; };
union FragB { uint4 u[2]; v16h v; };

__device__ __forceinline__ float fast_sigmoid(float x) {
    return 1.0f / (1.0f + __expf(-x));
}
__device__ __forceinline__ float fast_tanh(float x) {
    return 1.0f - 2.0f / (__expf(2.0f * x) + 1.0f);
}

// Native CDNA5 packed-f16 atomic add (no-return form, device scope) so the
// scatter phase cannot silently fall back to a CAS loop.
__device__ __forceinline__ void atomic_pk_add_f16(_Float16* addr, unsigned lo, unsigned hi) {
    asm volatile(
        "global_atomic_pk_add_f16 %0, %1, off scope:SCOPE_DEV\n\t"
        "global_atomic_pk_add_f16 %0, %2, off offset:4 scope:SCOPE_DEV"
        :
        : "v"(addr), "v"(lo), "v"(hi)
        : "memory");
}

// ---------------------------------------------------------------------------
// Kernel 1: weights fp32 -> f16 packed [512 x 256] (row n: W_ih[n,:] | W_hh[n,:])
//           bias = b_ih + b_hh
// ---------------------------------------------------------------------------
__global__ __launch_bounds__(256) void prep_kernel(
    const float* __restrict__ W_ih, const float* __restrict__ W_hh,
    const float* __restrict__ b_ih, const float* __restrict__ b_hh,
    _Float16* __restrict__ Wf16, float* __restrict__ bias)
{
    int idx = blockIdx.x * 256 + threadIdx.x;      // 0 .. 512*256-1
    int n = idx >> 8;
    int k = idx & 255;
    float v = (k < DM) ? W_ih[n * DM + k] : W_hh[n * DM + (k - DM)];
    Wf16[idx] = (_Float16)v;
    if (idx < NGATE) bias[idx] = b_ih[idx] + b_hh[idx];
}

// ---------------------------------------------------------------------------
// Kernel 2: COO SpMM scatter-add into f16 msg via packed f16 atomics.
// One wave per nonzero; each lane covers 4 columns (float4 gather of x_v).
// ---------------------------------------------------------------------------
__global__ __launch_bounds__(256) void spmm_kernel(
    const float* __restrict__ x_v, const float* __restrict__ edge_val,
    const int* __restrict__ edge_row, const int* __restrict__ edge_col,
    _Float16* __restrict__ msg)
{
    int t    = blockIdx.x * 256 + threadIdx.x;
    int nz   = t >> 5;
    int lane = t & 31;
    if (nz >= NNZ) return;

    int   row = edge_row[nz];
    int   col = edge_col[nz];
    float val = edge_val[nz];

    const float4 xv = *(const float4*)(x_v + (size_t)col * DM + lane * 4);

    union { _Float16 h[2]; unsigned u; } p0, p1;
    p0.h[0] = (_Float16)(xv.x * val);
    p0.h[1] = (_Float16)(xv.y * val);
    p1.h[0] = (_Float16)(xv.z * val);
    p1.h[1] = (_Float16)(xv.w * val);

    atomic_pk_add_f16(msg + (size_t)row * DM + lane * 4, p0.u, p1.u);
}

// ---------------------------------------------------------------------------
// Kernel 3: fused GEMM (WMMA f16 -> f32) + LSTM cell epilogue.
// Block = 16 edge rows, 8 waves; wave w owns d-slice [w*16, w*16+16).
// Each wave accumulates 4 gate tiles (i,f,g,o) sharing one A fragment per
// K-step: 8 K-steps x 4 = 32 v_wmma_f32_16x16x32_f16 per wave.
// ---------------------------------------------------------------------------
__global__ __launch_bounds__(256) void gemm_lstm_kernel(
    const _Float16* __restrict__ msg, const float* __restrict__ h_e,
    const float* __restrict__ c_e, const _Float16* __restrict__ Wf16,
    const float* __restrict__ bias,
    float* __restrict__ h_out, float* __restrict__ c_out)
{
    __shared__ _Float16 A[16 * LDA];   // 16 rows x 256 K (padded), f16

    const int etile = blockIdx.x;
    const int tid   = threadIdx.x;

    // ---- Stage A-panel into LDS: cols 0..127 from msg (f16 copy),
    //      cols 128..255 from h_e (fp32 -> f16). 16 halves per thread. ----
    {
        int m  = tid >> 4;           // row 0..15
        int kg = tid & 15;           // 16-half chunk 0..15
        int e  = etile * 16 + m;
        union { uint4 u[2]; _Float16 h[16]; } buf;
        if (kg < 8) {
            const uint4* src = (const uint4*)(msg + (size_t)e * DM + kg * 16);
            buf.u[0] = src[0];
            buf.u[1] = src[1];
        } else {
            const float4* src = (const float4*)(h_e + (size_t)e * DM + (kg - 8) * 16);
            float tmp[16];
#pragma unroll
            for (int j = 0; j < 4; ++j) {
                float4 f = src[j];
                tmp[4 * j + 0] = f.x; tmp[4 * j + 1] = f.y;
                tmp[4 * j + 2] = f.z; tmp[4 * j + 3] = f.w;
            }
#pragma unroll
            for (int j = 0; j < 16; ++j) buf.h[j] = (_Float16)tmp[j];
        }
        uint4* dst = (uint4*)&A[m * LDA + kg * 16];
        dst[0] = buf.u[0];
        dst[1] = buf.u[1];
    }
    __syncthreads();

    const int wave  = tid >> 5;      // d-tile 0..7
    const int lane  = tid & 31;
    const int halfg = lane >> 4;     // lane group 0/1
    const int nidx  = lane & 15;     // N within tile / M row for A

    v8f acc[4] = {v8f{}, v8f{}, v8f{}, v8f{}};

    const _Float16* arow = &A[nidx * LDA];   // A row M = lane&15

#pragma unroll
    for (int kt = 0; kt < 8; ++kt) {
        const int k0 = kt * 32;

        // A fragment (16x32 f16): lane<16 -> K {k0..k0+7, k0+16..k0+23};
        //                         lane>=16 -> shifted by 8.
        FragA af;
        af.u[0] = *(const uint4*)&arow[k0 + halfg * 8];
        af.u[1] = *(const uint4*)&arow[k0 + 16 + halfg * 8];

#pragma unroll
        for (int gi = 0; gi < 4; ++gi) {
            // B fragment (32x16 f16): lane holds column N, K {k0..k0+7, k0+16..k0+23}, lane>=16 shifted by 8.
            const _Float16* bp =
                Wf16 + ((size_t)(gi * DM + wave * 16 + nidx)) * KTOT + k0 + halfg * 8;
            FragB bf;
            bf.u[0] = ((const uint4*)bp)[0];
            bf.u[1] = ((const uint4*)(bp + 16))[0];

            acc[gi] = __builtin_amdgcn_wmma_f32_16x16x32_f16(
                false, af.v, false, bf.v, (short)0, acc[gi], false, false);
        }
    }

    // ---- LSTM epilogue using C/D layout: VGPR r -> M = r + halfg*8, N = lane&15
    const int d  = wave * 16 + nidx;
    const float bi = bias[d];
    const float bf = bias[DM + d];
    const float bg = bias[2 * DM + d];
    const float bo = bias[3 * DM + d];

#pragma unroll
    for (int r = 0; r < 8; ++r) {
        int    e   = etile * 16 + r + halfg * 8;
        size_t off = (size_t)e * DM + d;

        float iv = fast_sigmoid(acc[0][r] + bi);
        float fv = fast_sigmoid(acc[1][r] + bf);
        float gv = fast_tanh   (acc[2][r] + bg);
        float ov = fast_sigmoid(acc[3][r] + bo);

        float cn = fv * c_e[off] + iv * gv;
        float hn = ov * fast_tanh(cn);

        h_out[off] = hn;
        c_out[off] = cn;
    }
}

// ---------------------------------------------------------------------------
extern "C" void kernel_launch(void* const* d_in, const int* in_sizes, int n_in,
                              void* d_out, int out_size, void* d_ws, size_t ws_size,
                              hipStream_t stream)
{
    const float* x_v      = (const float*)d_in[0];
    const float* h_e      = (const float*)d_in[1];
    const float* c_e      = (const float*)d_in[2];
    const float* edge_val = (const float*)d_in[3];
    const int*   edge_row = (const int*)d_in[4];
    const int*   edge_col = (const int*)d_in[5];
    const float* W_ih     = (const float*)d_in[6];
    const float* W_hh     = (const float*)d_in[7];
    const float* b_ih     = (const float*)d_in[8];
    const float* b_hh     = (const float*)d_in[9];

    float* h_out = (float*)d_out;
    float* c_out = h_out + (size_t)NUM_E * DM;

    // Workspace layout: msg f16 | Wf16 | bias
    char*  ws        = (char*)d_ws;
    size_t msg_bytes = (size_t)NUM_E * DM * sizeof(_Float16);        // 102.4 MB
    _Float16* msg  = (_Float16*)ws;
    _Float16* Wf16 = (_Float16*)(ws + ((msg_bytes + 255) & ~(size_t)255));
    float*    bias = (float*)((char*)Wf16 + (size_t)NGATE * KTOT * sizeof(_Float16));

    hipMemsetAsync(msg, 0, msg_bytes, stream);

    prep_kernel<<<(NGATE * KTOT) / 256, 256, 0, stream>>>(W_ih, W_hh, b_ih, b_hh, Wf16, bias);
    spmm_kernel<<<NNZ / 8, 256, 0, stream>>>(x_v, edge_val, edge_row, edge_col, msg);
    gemm_lstm_kernel<<<NUM_E / 16, 256, 0, stream>>>(msg, h_e, c_e, Wf16, bias, h_out, c_out);
}
